// Informer_44143673868321
// MI455X (gfx1250) — matrix operand order repaired and checked by reading, not compiled, not run on hardware
//
#include <hip/hip_runtime.h>
#include <math.h>

typedef __attribute__((ext_vector_type(16))) _Float16 v16h;
typedef __attribute__((ext_vector_type(8)))  _Float16 half8;
typedef __attribute__((ext_vector_type(4)))  _Float16 half4;
typedef __attribute__((ext_vector_type(8)))  float    v8f;

#define BQ 8           // batch
#define NH 8           // heads
#define DM 512         // d_model
#define DH 64          // head dim
#define DFF 2048

#define BM 128
#define BN 128
#define BK 64
#define BKP 72         // padded LDS row (halves) to break bank conflicts; 72*2=144B (16B aligned)

// ---------------------------------------------------------------------------
// Staging helpers: global fp32 -> packed f16 registers -> LDS
// Requirements (hold for every call site): M%128==0, K%64==0, N%16==0.
// Only N may be partially covered by a 128-wide tile (N=64 case): handled
// with clamped addresses + value select (no exec-masked loads).
// ---------------------------------------------------------------------------
__device__ __forceinline__ void stage_load(
    const float* __restrict__ Ag, const float* __restrict__ Bg,
    int N, int lda, int ldb, int bm, int bn, int k0, int tid, int flags,
    half4* ra, half4* rb)
{
#pragma unroll
    for (int i = 0; i < 8; ++i) {               // A tile: 128 x 64 fp32
        int idx4 = tid + i * 256;
        int r = idx4 >> 4, c = (idx4 & 15) << 2;
        const float4 v = *(const float4*)&Ag[(long)(bm + r) * lda + k0 + c];
        half4 h; h[0] = (_Float16)v.x; h[1] = (_Float16)v.y;
        h[2] = (_Float16)v.z; h[3] = (_Float16)v.w;
        ra[i] = h;
    }
    if (flags & 1) {                            // B is [N][K]
#pragma unroll
        for (int i = 0; i < 8; ++i) {
            int idx4 = tid + i * 256;
            int n = idx4 >> 4, c = (idx4 & 15) << 2;
            int gn = bn + n;
            int gc = (gn < N) ? gn : 0;
            const float4 v = *(const float4*)&Bg[(long)gc * ldb + k0 + c];
            _Float16 z = (_Float16)0.0f;
            bool ok = gn < N;
            half4 h;
            h[0] = ok ? (_Float16)v.x : z; h[1] = ok ? (_Float16)v.y : z;
            h[2] = ok ? (_Float16)v.z : z; h[3] = ok ? (_Float16)v.w : z;
            rb[i] = h;
        }
    } else {                                    // B is [K][N]
#pragma unroll
        for (int i = 0; i < 8; ++i) {
            int idx4 = tid + i * 256;
            int k = idx4 >> 5, c = (idx4 & 31) << 2;
            int gn = bn + c;
            int gc = (gn < N) ? gn : 0;         // N%4==0 -> whole vec uniform
            const float4 v = *(const float4*)&Bg[(long)(k0 + k) * ldb + gc];
            _Float16 z = (_Float16)0.0f;
            bool ok = gn < N;
            half4 h;
            h[0] = ok ? (_Float16)v.x : z; h[1] = ok ? (_Float16)v.y : z;
            h[2] = ok ? (_Float16)v.z : z; h[3] = ok ? (_Float16)v.w : z;
            rb[i] = h;
        }
    }
}

__device__ __forceinline__ void stage_store(
    _Float16* __restrict__ sA, _Float16* __restrict__ sB,
    int tid, int flags, const half4* ra, const half4* rb)
{
#pragma unroll
    for (int i = 0; i < 8; ++i) {
        int idx4 = tid + i * 256;
        int r = idx4 >> 4, c = (idx4 & 15) << 2;
        *(half4*)&sA[r * BKP + c] = ra[i];
    }
    if (flags & 1) {
#pragma unroll
        for (int i = 0; i < 8; ++i) {
            int idx4 = tid + i * 256;
            int n = idx4 >> 4, c = (idx4 & 15) << 2;
            *(half4*)&sB[n * BKP + c] = rb[i];
        }
    } else {                                    // transpose scatter [K][N]->[n][k]
#pragma unroll
        for (int i = 0; i < 8; ++i) {
            int idx4 = tid + i * 256;
            int k = idx4 >> 5, c = (idx4 & 31) << 2;
#pragma unroll
            for (int j = 0; j < 4; ++j) sB[(c + j) * BKP + k] = rb[i][j];
        }
    }
}

// ---------------------------------------------------------------------------
// Generic batched WMMA GEMM:  C = alpha * A @ B (+bias)(+res)(+gelu)(+=C)
// flags: 1=transB([N][K]), 2=gelu, 4=accumulate into C
// Double-buffered LDS, BK=64 => 16 v_wmma per barrier.
// ---------------------------------------------------------------------------
__global__ __launch_bounds__(256) void wmma_gemm(
    const float* __restrict__ A, const float* __restrict__ Bm, float* __restrict__ C,
    int M, int N, int K, int lda, int ldb, int ldc,
    long sA, long sB, long sC,
    const float* __restrict__ bias, const float* __restrict__ res,
    float alpha, int flags)
{
    __shared__ _Float16 shA[2][BM * BKP];
    __shared__ _Float16 shB[2][BN * BKP];

    const int tid = threadIdx.x;
    const int bm  = blockIdx.y * BM;
    const int bn  = blockIdx.x * BN;
    const int bz  = blockIdx.z;

    const float* Ag = A  + (long)bz * sA;
    const float* Bg = Bm + (long)bz * sB;
    float*       Cg = C  + (long)bz * sC;
    const float* Rg = res ? (res + (long)bz * sC) : nullptr;

    const int wid   = tid >> 5;
    const int lane  = tid & 31;
    const int wm    = (wid & 1) * 64;   // 2 waves over M
    const int wn    = (wid >> 1) * 32;  // 4 waves over N
    const int lhalf = lane >> 4;
    const int l15   = lane & 15;

    v8f acc[4][2];
#pragma unroll
    for (int mt = 0; mt < 4; ++mt)
#pragma unroll
        for (int nt = 0; nt < 2; ++nt)
#pragma unroll
            for (int i = 0; i < 8; ++i) acc[mt][nt][i] = 0.0f;

    half4 ra[8], rb[8];
    stage_load(Ag, Bg, N, lda, ldb, bm, bn, 0, tid, flags, ra, rb);
    stage_store(shA[0], shB[0], tid, flags, ra, rb);
    __syncthreads();

    const int nk = K >> 6;              // K % 64 == 0
    for (int kt = 0; kt < nk; ++kt) {
        const int cur  = kt & 1;
        const bool more = (kt + 1) < nk;
        if (more)
            stage_load(Ag, Bg, N, lda, ldb, bm, bn, (kt + 1) << 6, tid, flags, ra, rb);

        const _Float16* sAc = shA[cur];
        const _Float16* sBc = shB[cur];
#pragma unroll
        for (int ks = 0; ks < 2; ++ks) {
            const int kb = ks * 32;
            v16h af[4], bf[2];
#pragma unroll
            for (int mt = 0; mt < 4; ++mt) {
                int row = wm + mt * 16 + l15;
                half8 lo = *(const half8*)&sAc[row * BKP + kb + lhalf * 8];
                half8 hi = *(const half8*)&sAc[row * BKP + kb + 16 + lhalf * 8];
#pragma unroll
                for (int i = 0; i < 8; ++i) { af[mt][i] = lo[i]; af[mt][8 + i] = hi[i]; }
            }
#pragma unroll
            for (int nt = 0; nt < 2; ++nt) {
                int coln = wn + nt * 16 + l15;
                half8 b0 = *(const half8*)&sBc[coln * BKP + kb + lhalf * 8];
                half8 b1 = *(const half8*)&sBc[coln * BKP + kb + 16 + lhalf * 8];
#pragma unroll
                for (int i = 0; i < 8; ++i) { bf[nt][i] = b0[i]; bf[nt][8 + i] = b1[i]; }
            }
#pragma unroll
            for (int mt = 0; mt < 4; ++mt)
#pragma unroll
                for (int nt = 0; nt < 2; ++nt)
                    acc[mt][nt] = __builtin_amdgcn_wmma_f32_16x16x32_f16(
                        false, af[mt], false, bf[nt], (short)0, acc[mt][nt], false, false);
        }
        if (more)
            stage_store(shA[cur ^ 1], shB[cur ^ 1], tid, flags, ra, rb);
        __syncthreads();
    }

    // ---- epilogue (rows always valid: M%128==0; col-tile check is uniform) ----
#pragma unroll
    for (int mt = 0; mt < 4; ++mt)
#pragma unroll
        for (int nt = 0; nt < 2; ++nt) {
            int colbase = bn + wn + nt * 16;
            if (colbase >= N) continue;         // N%16==0 -> uniform over tile
            int col = colbase + l15;
            float bv = bias ? bias[col] : 0.0f;
#pragma unroll
            for (int r = 0; r < 8; ++r) {
                int row = bm + wm + mt * 16 + r + lhalf * 8;
                float v = acc[mt][nt][r] * alpha + bv;
                long off = (long)row * ldc + col;
                if (Rg) v += Rg[off];
                if (flags & 2) v = 0.5f * v * (1.0f + erff(v * 0.70710678118f));
                if (flags & 4) v += Cg[off];
                Cg[off] = v;
            }
        }
}

// ---------------------------------------------------------------------------
// Elementwise / reduction kernels
// ---------------------------------------------------------------------------
__device__ __forceinline__ float pe_val(int pos, int d) {
    const float Cc = 0.017988946039015984f; // ln(10000)/512
    float div = expf(-(float)(d & ~1) * Cc);
    float a = (float)pos * div;
    return (d & 1) ? cosf(a) : sinf(a);
}

__global__ __launch_bounds__(128) void k_embed(
    const float* __restrict__ x, const int* __restrict__ mark,
    const float* __restrict__ w, float* __restrict__ out, int L)
{
    int bl = blockIdx.x;
    int b = bl / L, l = bl % L;
    int tid = threadIdx.x;
    __shared__ float xl[21];
    __shared__ int mk[4];
    if (tid < 21) {
        int t = tid / 7, c = tid % 7;
        int ll = (l + t - 1 + L) % L;
        xl[tid] = x[((long)b * L + ll) * 7 + c];
    }
    if (tid < 4) mk[tid] = mark[((long)b * L + l) * 4 + tid];
    __syncthreads();
#pragma unroll
    for (int uu = 0; uu < 4; ++uu) {
        int d = tid + uu * 128;
        float conv = 0.0f;
#pragma unroll
        for (int c = 0; c < 7; ++c)
#pragma unroll
            for (int t = 0; t < 3; ++t)
                conv += xl[t * 7 + c] * w[d * 21 + c * 3 + t];
        float v = conv + pe_val(l, d);
#pragma unroll
        for (int f = 0; f < 4; ++f) v += pe_val(mk[f], d);
        out[((long)b * L + l) * (long)DM + d] = v;
    }
}

__global__ __launch_bounds__(256) void k_layernorm(
    const float* __restrict__ in, float* __restrict__ out,
    const float* __restrict__ g, const float* __restrict__ b)
{
    int row = blockIdx.x, tid = threadIdx.x;
    const float* x = in + (long)row * DM;
    float v0 = x[tid], v1 = x[tid + 256];
    __shared__ float red[256];
    red[tid] = v0 + v1;
    __syncthreads();
    for (int s = 128; s > 0; s >>= 1) { if (tid < s) red[tid] += red[tid + s]; __syncthreads(); }
    float m = red[0] * (1.0f / DM);
    __syncthreads();
    float d0 = v0 - m, d1 = v1 - m;
    red[tid] = d0 * d0 + d1 * d1;
    __syncthreads();
    for (int s = 128; s > 0; s >>= 1) { if (tid < s) red[tid] += red[tid + s]; __syncthreads(); }
    float inv = rsqrtf(red[0] * (1.0f / DM) + 1e-5f);
    out[(long)row * DM + tid]       = d0 * inv * g[tid] + b[tid];
    out[(long)row * DM + tid + 256] = d1 * inv * g[tid + 256] + b[tid + 256];
}

// (B*L, 512) -> (B, H, L, 64)
__global__ __launch_bounds__(256) void k_pack(const float* __restrict__ in,
                                              float* __restrict__ out, int L, long total)
{
    long i = (long)blockIdx.x * 256 + threadIdx.x;
    if (i >= total) return;
    int e = (int)(i & 63);
    long r = i >> 6;
    int l = (int)(r % L); r /= L;
    int h = (int)(r % NH); int b = (int)(r / NH);
    out[i] = in[((long)b * L + l) * DM + h * DH + e];
}

// (B, H, L, 64) -> (B*L, 512)
__global__ __launch_bounds__(256) void k_unpack(const float* __restrict__ in,
                                                float* __restrict__ out, int L, long total)
{
    long i = (long)blockIdx.x * 256 + threadIdx.x;
    if (i >= total) return;
    int c = (int)(i & (DM - 1));
    long row = i >> 9;
    int b = (int)(row / L), l = (int)(row % L);
    int h = c >> 6, e = c & 63;
    out[i] = in[(((long)b * NH + h) * L + l) * DH + e];
}

// ProbSparse: sparsity measure M = max_j(Q.K_samp) - sum_j(Q.K_samp)/L over sampled keys
__global__ __launch_bounds__(256) void k_pa_m(const float* __restrict__ Q,
                                              const float* __restrict__ K,
                                              float* __restrict__ Mo,
                                              int L, int U, int layer)
{
    long i = (long)blockIdx.x * 256 + threadIdx.x;
    if (i >= (long)BQ * NH * L) return;
    long bh = i / L;
    int  l  = (int)(i % L);
    const float* q = Q + ((long)bh * L + l) * DH;
    float mx = -INFINITY, sm = 0.0f;
    for (int j = 0; j < U; ++j) {
        unsigned h = (unsigned)l * 2654435761u ^ (unsigned)j * 40503u ^ (unsigned)layer * 2246822519u;
        h ^= h >> 13; h *= 2654435761u; h ^= h >> 16;
        int ki = (int)(h % (unsigned)L);
        const float* kr = K + ((long)bh * L + ki) * DH;
        float d = 0.0f;
#pragma unroll
        for (int e = 0; e < DH; ++e) d += q[e] * kr[e];
        mx = fmaxf(mx, d); sm += d;
    }
    Mo[i] = mx - sm / (float)L;
}

__global__ __launch_bounds__(256) void k_pa_topk(const float* __restrict__ Mo,
                                                 int* __restrict__ top, int L, int u)
{
    int bh = blockIdx.x, tid = threadIdx.x;
    __shared__ float vals[1024];
    __shared__ float rv[256];
    __shared__ int   ri[256];
    for (int i = tid; i < L; i += 256) vals[i] = Mo[(long)bh * L + i];
    __syncthreads();
    for (int t = 0; t < u; ++t) {
        float bv = -INFINITY; int bi = 0;
        for (int i = tid; i < L; i += 256)
            if (vals[i] > bv) { bv = vals[i]; bi = i; }
        rv[tid] = bv; ri[tid] = bi;
        __syncthreads();
        for (int s = 128; s > 0; s >>= 1) {
            if (tid < s && rv[tid + s] > rv[tid]) { rv[tid] = rv[tid + s]; ri[tid] = ri[tid + s]; }
            __syncthreads();
        }
        if (tid == 0) { top[bh * u + t] = ri[0]; vals[ri[0]] = -INFINITY; }
        __syncthreads();
    }
}

__global__ __launch_bounds__(64) void k_pa_vmean(const float* __restrict__ V,
                                                 float* __restrict__ vm, int S)
{
    int bh = blockIdx.x, e = threadIdx.x;
    float a = 0.0f;
    for (int j = 0; j < S; ++j) a += V[((long)bh * S + j) * DH + e];
    vm[bh * DH + e] = a / (float)S;
}

__global__ __launch_bounds__(256) void k_pa_fill(const float* __restrict__ vm,
                                                 float* __restrict__ ctx, int L, long total)
{
    long i = (long)blockIdx.x * 256 + threadIdx.x;
    if (i >= total) return;
    int e = (int)(i & 63);
    long bh = i / ((long)L * DH);
    ctx[i] = vm[bh * DH + e];
}

__global__ __launch_bounds__(256) void k_pa_update(const float* __restrict__ Q,
                                                   const float* __restrict__ K,
                                                   const float* __restrict__ V,
                                                   const int* __restrict__ top,
                                                   float* __restrict__ ctx,
                                                   int S, int u)
{
    int bh = blockIdx.x / u, t = blockIdx.x % u;
    int tid = threadIdx.x;
    int qi = top[bh * u + t];
    __shared__ float sc[1024];
    __shared__ float red[256];
    __shared__ float q[DH];
    if (tid < DH) q[tid] = Q[((long)bh * S + qi) * DH + tid];
    __syncthreads();
    for (int j = tid; j < S; j += 256) {
        const float* kr = K + ((long)bh * S + j) * DH;
        float d = 0.0f;
#pragma unroll
        for (int e = 0; e < DH; ++e) d += q[e] * kr[e];
        sc[j] = d * 0.125f;
    }
    __syncthreads();
    float mx = -INFINITY;
    for (int j = tid; j < S; j += 256) mx = fmaxf(mx, sc[j]);
    red[tid] = mx; __syncthreads();
    for (int s = 128; s > 0; s >>= 1) { if (tid < s) red[tid] = fmaxf(red[tid], red[tid + s]); __syncthreads(); }
    mx = red[0]; __syncthreads();
    float sm = 0.0f;
    for (int j = tid; j < S; j += 256) { float p = expf(sc[j] - mx); sc[j] = p; sm += p; }
    red[tid] = sm; __syncthreads();
    for (int s = 128; s > 0; s >>= 1) { if (tid < s) red[tid] += red[tid + s]; __syncthreads(); }
    float inv = 1.0f / red[0];
    __syncthreads();
    if (tid < DH) {
        float a = 0.0f;
        for (int j = 0; j < S; ++j) a += sc[j] * V[((long)bh * S + j) * DH + tid];
        ctx[((long)bh * S + qi) * DH + tid] = a * inv;
    }
}

__global__ __launch_bounds__(256) void k_softmax(float* __restrict__ sc, int Lq, int S, int causal)
{
    long row = blockIdx.x;
    int l = (int)(row % Lq);
    float* p = sc + row * (long)S;
    int limit = causal ? (l + 1) : S;
    int tid = threadIdx.x;
    __shared__ float red[256];
    float mx = -INFINITY;
    for (int j = tid; j < limit; j += 256) mx = fmaxf(mx, p[j]);
    red[tid] = mx; __syncthreads();
    for (int s = 128; s > 0; s >>= 1) { if (tid < s) red[tid] = fmaxf(red[tid], red[tid + s]); __syncthreads(); }
    mx = red[0]; __syncthreads();
    float sm = 0.0f;
    for (int j = tid; j < S; j += 256) {
        float v = (j < limit) ? expf(p[j] - mx) : 0.0f;
        p[j] = v; sm += v;
    }
    red[tid] = sm; __syncthreads();
    for (int s = 128; s > 0; s >>= 1) { if (tid < s) red[tid] += red[tid + s]; __syncthreads(); }
    float inv = 1.0f / red[0];
    for (int j = tid; j < S; j += 256) p[j] *= inv;
}

// circular pad over time: (B,L,512) -> (B,L+2,512)
__global__ __launch_bounds__(256) void k_xpad(const float* __restrict__ x,
                                              float* __restrict__ xp, int L, long total)
{
    long i = (long)blockIdx.x * 256 + threadIdx.x;
    if (i >= total) return;
    int c = (int)(i & (DM - 1));
    long r = i >> 9;
    int lp = (int)(r % (L + 2));
    int b  = (int)(r / (L + 2));
    int l  = (lp - 1 + L) % L;
    xp[i] = x[((long)b * L + l) * DM + c];
}

// (O,I,3) -> wt[t][i][o]
__global__ __launch_bounds__(256) void k_repack_convw(const float* __restrict__ w,
                                                      float* __restrict__ wt)
{
    long i = (long)blockIdx.x * 256 + threadIdx.x;
    if (i >= (long)3 * DM * DM) return;
    int t = (int)(i / (DM * DM));
    int r = (int)(i % (DM * DM));
    int ii = r / DM, o = r % DM;
    wt[i] = w[(long)o * (DM * 3) + ii * 3 + t];
}

__global__ __launch_bounds__(256) void k_bn_stats(const float* __restrict__ y,
                                                  float* __restrict__ mean,
                                                  float* __restrict__ var, int n)
{
    int d = blockIdx.x, tid = threadIdx.x;
    __shared__ float rs[256], rq[256];
    float s = 0.0f, q = 0.0f;
    for (int r = tid; r < n; r += 256) {
        float v = y[(long)r * DM + d];
        s += v; q += v * v;
    }
    rs[tid] = s; rq[tid] = q;
    __syncthreads();
    for (int st = 128; st > 0; st >>= 1) {
        if (tid < st) { rs[tid] += rs[tid + st]; rq[tid] += rq[tid + st]; }
        __syncthreads();
    }
    if (tid == 0) {
        float m = rs[0] / (float)n;
        mean[d] = m;
        var[d]  = rq[0] / (float)n - m * m;
    }
}

__global__ __launch_bounds__(256) void k_pool_bn_elu(
    const float* __restrict__ y, const float* __restrict__ mean, const float* __restrict__ var,
    const float* __restrict__ g, const float* __restrict__ bb,
    float* __restrict__ out, int L, long total)
{
    long i = (long)blockIdx.x * 256 + threadIdx.x;
    if (i >= total) return;
    int d = (int)(i & (DM - 1));
    long r = i >> 9;
    int Lh = L / 2;
    int lp = (int)(r % Lh);
    int b  = (int)(r / Lh);
    float m = mean[d], inv = rsqrtf(var[d] + 1e-5f), ga = g[d], be = bb[d];
    float best = -INFINITY;
#pragma unroll
    for (int t = 0; t < 3; ++t) {
        int l = 2 * lp + t - 1;
        if (l < 0 || l >= L) continue;
        float v = (y[((long)b * L + l) * DM + d] - m) * inv * ga + be;
        v = v > 0.0f ? v : expf(v) - 1.0f;
        best = fmaxf(best, v);
    }
    out[i] = best;
}

__global__ __launch_bounds__(256) void k_proj_out(const float* __restrict__ x,
                                                  const float* __restrict__ w,
                                                  const float* __restrict__ bias,
                                                  float* __restrict__ out)
{
    long i = (long)blockIdx.x * 256 + threadIdx.x;
    if (i >= (long)BQ * 256 * 7) return;
    int c = (int)(i % 7);
    long r = i / 7;
    int l = (int)(r % 256), b = (int)(r / 256);
    const float* xr = x + ((long)b * 512 + 256 + l) * DM;
    float a = bias[c];
    for (int k = 0; k < DM; ++k) a += xr[k] * w[k * 7 + c];
    out[i] = a;
}

// ---------------------------------------------------------------------------
// Host orchestration
// ---------------------------------------------------------------------------
extern "C" void kernel_launch(void* const* d_in, const int* in_sizes, int n_in,
                              void* d_out, int out_size, void* d_ws, size_t ws_size,
                              hipStream_t stream)
{
    (void)in_sizes; (void)n_in; (void)out_size; (void)ws_size;

    const float* x_enc    = (const float*)d_in[0];
    const int*   mark_enc = (const int*)d_in[1];
    const float* x_dec    = (const float*)d_in[2];
    const int*   mark_dec = (const int*)d_in[3];
    auto P = [&](int i) { return (const float*)d_in[4 + i]; };

    // pytree (sorted-key) param offsets
    const int DECN_B = 52, DECN_G = 53, DEC_TOK = 54;
    auto CV = [&](int i) { return 55 + 4 * i; };   // b, bn_b, bn_g, w
    auto EL = [&](int i) { return 63 + 16 * i; };  // k.b,k.w,o.b,o.w,q.b,q.w,v.b,v.w,b1,b2,n1.b,n1.g,n2.b,n2.g,w1,w2
    const int ENCN_B = 111, ENCN_G = 112, ENC_TOK = 113, PROJ_B = 114, PROJ_W = 115;

    const long ACT = (long)BQ * 1024 * DM;
    float* base = (float*)d_ws;
    long off = 0;
    auto alloc = [&](long n) { float* p = base + off; off += n; return p; };
    float* fX   = alloc(ACT);
    float* fT   = alloc(ACT);
    float* fQ   = alloc(ACT);
    float* fK   = alloc(ACT);
    float* fV   = alloc(ACT);
    float* fCTX = alloc(ACT);
    float* fSC  = alloc((long)16777216);  // scores / FFN hidden (aliased)
    float* fENC = alloc((long)BQ * 256 * DM);
    float* fXP  = alloc((long)BQ * 1026 * DM);
    float* fWT  = alloc((long)3 * DM * DM);
    float* fM   = alloc((long)BQ * NH * 1024);
    int*   iTOP = (int*)alloc(BQ * NH * 64);
    float* fVM  = alloc(BQ * NH * DH);
    float* fBNm = alloc(DM);
    float* fBNv = alloc(DM);

    auto gemm = [&](const float* A, const float* Bm, float* C,
                    int M, int N, int K, int lda, int ldb, int ldc,
                    long sA, long sB, long sC, int batch,
                    const float* bias, const float* res, float alpha, int flags) {
        dim3 g((N + BN - 1) / BN, (M + BM - 1) / BM, batch);
        wmma_gemm<<<g, 256, 0, stream>>>(A, Bm, C, M, N, K, lda, ldb, ldc,
                                         sA, sB, sC, bias, res, alpha, flags);
    };
    auto lnorm = [&](const float* in, float* out, const float* g, const float* b, int rows) {
        k_layernorm<<<rows, 256, 0, stream>>>(in, out, g, b);
    };
    auto pack = [&](const float* in, float* out, int L) {
        long tot = (long)BQ * L * DM;
        k_pack<<<(unsigned)((tot + 255) / 256), 256, 0, stream>>>(in, out, L, tot);
    };
    auto unpack = [&](const float* in, float* out, int L) {
        long tot = (long)BQ * L * DM;
        k_unpack<<<(unsigned)((tot + 255) / 256), 256, 0, stream>>>(in, out, L, tot);
    };
    auto linproj = [&](const float* in, const float* w, const float* b, float* out,
                       int M, const float* res, int flags, int N, int K) {
        gemm(in, w, out, M, N, K, K, N, N, 0, 0, 0, 1, b, res, 1.0f, flags);
    };

    // ---------------- ProbSparse encoder layer ----------------
    auto enc_layer = [&](int eb, int L, int layer_id) {
        int M = BQ * L;
        linproj(fX, P(eb + 5), P(eb + 4), fT, M, nullptr, 0, DM, DM); pack(fT, fQ, L);
        linproj(fX, P(eb + 1), P(eb + 0), fT, M, nullptr, 0, DM, DM); pack(fT, fK, L);
        linproj(fX, P(eb + 7), P(eb + 6), fT, M, nullptr, 0, DM, DM); pack(fT, fV, L);

        int ce = (int)ceil(log((double)L));
        int u = 5 * ce; if (u > L) u = L;
        long nq = (long)BQ * NH * L;
        k_pa_m<<<(unsigned)((nq + 255) / 256), 256, 0, stream>>>(fQ, fK, fM, L, u, layer_id);
        k_pa_topk<<<BQ * NH, 256, 0, stream>>>(fM, iTOP, L, u);
        k_pa_vmean<<<BQ * NH, 64, 0, stream>>>(fV, fVM, L);
        long tot = nq * DH;
        k_pa_fill<<<(unsigned)((tot + 255) / 256), 256, 0, stream>>>(fVM, fCTX, L, tot);
        k_pa_update<<<BQ * NH * u, 256, 0, stream>>>(fQ, fK, fV, iTOP, fCTX, L, u);
        unpack(fCTX, fT, L);
        linproj(fT, P(eb + 3), P(eb + 2), fX, M, fX, 0, DM, DM);       // O-proj + residual
        lnorm(fX, fX, P(eb + 11), P(eb + 10), M);                      // norm1
        gemm(fX, P(eb + 14), fSC, M, DFF, DM, DM, DFF, DFF, 0, 0, 0, 1,
             P(eb + 8), nullptr, 1.0f, 2);                             // gelu(x@w1+b1)
        gemm(fSC, P(eb + 15), fX, M, DM, DFF, DFF, DM, DM, 0, 0, 0, 1,
             P(eb + 9), fX, 1.0f, 0);                                  // @w2+b2 + residual
        lnorm(fX, fX, P(eb + 13), P(eb + 12), M);                      // norm2
    };

    // ---------------- distilling conv layer ----------------
    auto conv_layer = [&](int cb, int L) {
        long tp = (long)BQ * (L + 2) * DM;
        k_xpad<<<(unsigned)((tp + 255) / 256), 256, 0, stream>>>(fX, fXP, L, tp);
        long tw = (long)3 * DM * DM;
        k_repack_convw<<<(unsigned)((tw + 255) / 256), 256, 0, stream>>>(P(cb + 3), fWT);
        for (int t = 0; t < 3; ++t)
            gemm(fXP + (long)t * DM, fWT + (long)t * DM * DM, fT,
                 L, DM, DM, DM, DM, DM,
                 (long)(L + 2) * DM, 0, (long)L * DM, BQ,
                 (t == 0) ? P(cb + 0) : nullptr, nullptr, 1.0f, (t == 0) ? 0 : 4);
        k_bn_stats<<<DM, 256, 0, stream>>>(fT, fBNm, fBNv, BQ * L);
        long to = (long)BQ * (L / 2) * DM;
        k_pool_bn_elu<<<(unsigned)((to + 255) / 256), 256, 0, stream>>>(
            fT, fBNm, fBNv, P(cb + 2), P(cb + 1), fX, L, to);
    };

    // ---------------- full attention (decoder) ----------------
    auto full_attn = [&](const float* qw, const float* qb, const float* kw, const float* kb,
                         const float* vw, const float* vb, const float* ow, const float* ob,
                         const float* kvsrc, int Lq, int S, bool causal) {
        linproj(fX, qw, qb, fT, BQ * Lq, nullptr, 0, DM, DM); pack(fT, fQ, Lq);
        linproj(kvsrc, kw, kb, fT, BQ * S, nullptr, 0, DM, DM); pack(fT, fK, S);
        linproj(kvsrc, vw, vb, fT, BQ * S, nullptr, 0, DM, DM); pack(fT, fV, S);
        // scores = Q @ K^T / sqrt(64)   (batched over B*H)
        gemm(fQ, fK, fSC, Lq, S, DH, DH, DH, S,
             (long)Lq * DH, (long)S * DH, (long)Lq * S, BQ * NH,
             nullptr, nullptr, 0.125f, 1);
        k_softmax<<<BQ * NH * Lq, 256, 0, stream>>>(fSC, Lq, S, causal ? 1 : 0);
        // ctx = A @ V
        gemm(fSC, fV, fCTX, Lq, DH, S, S, DH, DH,
             (long)Lq * S, (long)S * DH, (long)Lq * DH, BQ * NH,
             nullptr, nullptr, 1.0f, 0);
        unpack(fCTX, fT, Lq);
        linproj(fT, ow, ob, fX, BQ * Lq, fX, 0, DM, DM);
    };

    // =================== encoder ===================
    k_embed<<<BQ * 1024, 128, 0, stream>>>(x_enc, mark_enc, P(ENC_TOK), fX, 1024);
    enc_layer(EL(0), 1024, 0);
    conv_layer(CV(0), 1024);
    enc_layer(EL(1), 512, 1);
    conv_layer(CV(1), 512);
    enc_layer(EL(2), 256, 2);
    lnorm(fX, fENC, P(ENCN_G), P(ENCN_B), BQ * 256);

    // =================== decoder ===================
    k_embed<<<BQ * 512, 128, 0, stream>>>(x_dec, mark_dec, P(DEC_TOK), fX, 512);
    for (int i = 0; i < 2; ++i) {
        int DL = 26 * i;
        full_attn(P(DL + 21), P(DL + 20), P(DL + 17), P(DL + 16),
                  P(DL + 23), P(DL + 22), P(DL + 19), P(DL + 18),
                  fX, 512, 512, true);                                  // masked self-attn
        lnorm(fX, fX, P(DL + 11), P(DL + 10), BQ * 512);                // norm1
        full_attn(P(DL + 7), P(DL + 6), P(DL + 3), P(DL + 2),
                  P(DL + 9), P(DL + 8), P(DL + 5), P(DL + 4),
                  fENC, 512, 256, false);                               // cross-attn
        lnorm(fX, fX, P(DL + 13), P(DL + 12), BQ * 512);                // norm2
        gemm(fX, P(DL + 24), fSC, BQ * 512, DFF, DM, DM, DFF, DFF, 0, 0, 0, 1,
             P(DL + 0), nullptr, 1.0f, 2);
        gemm(fSC, P(DL + 25), fX, BQ * 512, DM, DFF, DFF, DM, DM, 0, 0, 0, 1,
             P(DL + 1), fX, 1.0f, 0);
        lnorm(fX, fX, P(DL + 15), P(DL + 14), BQ * 512);                // norm3
    }
    lnorm(fX, fX, P(DECN_G), P(DECN_B), BQ * 512);

    long to = (long)BQ * 256 * 7;
    k_proj_out<<<(unsigned)((to + 255) / 256), 256, 0, stream>>>(
        fX, P(PROJ_W), P(PROJ_B), (float*)d_out);
}
